// Decoder_30185030156342
// MI455X (gfx1250) — matrix operand order repaired and checked by reading, not compiled, not run on hardware
//
#include <hip/hip_runtime.h>
#include <hip/hip_bf16.h>

// ---------------- problem constants ----------------
#define BB   32768   // batch
#define EE   64      // embedding dim
#define HH   128     // hidden dim
#define TT   12      // pred len
#define NG   512     // 4*H gate width
#define KP   192     // K stride (halfs) for LDS tiles (16B-aligned rows)

#define THREADS 256          // 8 wave32
#define ROWS_PER_WAVE 16
#define ROWS_PER_WG   128    // 8 waves * 16 rows
#define NUM_WG (BB / ROWS_PER_WG)

// ---------------- LDS layout (dynamic) ----------------
#define WT_BYTES   (NG * KP * 2)                    // 196608 : f16 weights [n][k]
#define XH_BYTES   (8 * 2 * ROWS_PER_WAVE * KP * 2) // 98304  : per-wave ping-pong x||h tiles
#define BL_BYTES   (NG * 4)                         // 2048   : b_ih + b_hh (f32)
#define WSE_BYTES  (EE * 4 * 4)                     // 1024   : [e]{w0,w1,w2,b}
#define WHP_BYTES  (2 * HH * 4)                     // 1024
#define BHP_BYTES  (16)
#define RB_BYTES   (8 * 16 * 4 * 4)                 // 2048   : per-wave {rel0,rel1,spd,_}
#define SMEM_BYTES (WT_BYTES + XH_BYTES + BL_BYTES + WSE_BYTES + WHP_BYTES + BHP_BYTES + RB_BYTES)

typedef __attribute__((ext_vector_type(16))) _Float16 v16h;
typedef __attribute__((ext_vector_type(8)))  _Float16 v8h;
typedef __attribute__((ext_vector_type(8)))  float    v8f;

#define WMMA_F16(a, b, c) \
  __builtin_amdgcn_wmma_f32_16x16x32_f16(false, (a), false, (b), (short)0, (c), false, false)

// Native TRANS-unit tanh on gfx1250 (co-executes with XDL WMMA); exp fallback otherwise.
#if __has_builtin(__builtin_amdgcn_tanhf)
__device__ __forceinline__ float tanh_(float x) { return __builtin_amdgcn_tanhf(x); }
#else
__device__ __forceinline__ float tanh_(float x) {
  float e = __expf(-2.0f * x);
  return (1.0f - e) / (1.0f + e);
}
#endif
// sigmoid(x) = 0.5*tanh(x/2) + 0.5 ; callers pre-scale by 0.5
__device__ __forceinline__ float sig_half_(float xh) { return __builtin_fmaf(0.5f, tanh_(xh), 0.5f); }

// A fragment (16x32 f16): lane<16 -> M=lane, halfs = K {0..7, 16..23}; lane>=16 -> +8
__device__ __forceinline__ v16h load_frag_a(const _Float16* xh, int lane, int kt) {
  int m = lane & 15, hi = lane >> 4;
  const _Float16* p = xh + m * KP + kt * 32 + hi * 8;
  v8h c0 = *(const v8h*)p;
  v8h c1 = *(const v8h*)(p + 16);
  v16h a;
#pragma unroll
  for (int i = 0; i < 8; ++i) { a[i] = c0[i]; a[8 + i] = c1[i]; }
  return a;
}

// B fragment (32x16 f16): lane&15 -> N column, halfs = K {0..7, 16..23}; lane>=16 -> +8
__device__ __forceinline__ v16h load_frag_b(const _Float16* wt, int lane, int kt, int nt) {
  int lo = lane & 15, hi = lane >> 4;
  const _Float16* p = wt + (nt * 16 + lo) * KP + kt * 32 + hi * 8;
  v8h c0 = *(const v8h*)p;
  v8h c1 = *(const v8h*)(p + 16);
  v16h b;
#pragma unroll
  for (int i = 0; i < 8; ++i) { b[i] = c0[i]; b[8 + i] = c1[i]; }
  return b;
}

__device__ __forceinline__ void embed_x(_Float16* xh, const float* rb, const float* wse4,
                                        int lane) {
  int m = lane & 15;
  int elo = (lane >> 4) * 32;
  float r0 = rb[m * 4 + 0], r1 = rb[m * 4 + 1], sp = rb[m * 4 + 2];
#pragma unroll 8
  for (int e = elo; e < elo + 32; ++e) {
    float v = r0 * wse4[e * 4 + 0] + r1 * wse4[e * 4 + 1] + sp * wse4[e * 4 + 2] +
              wse4[e * 4 + 3];
    xh[m * KP + e] = (_Float16)v;
  }
}

__global__ __launch_bounds__(THREADS, 1) void decoder_lstm_wmma(
    const float* __restrict__ last_pos,       // unused by outputs
    const float* __restrict__ last_pos_rel,   // [B,2]
    const float* __restrict__ h0,             // [B,H]
    const float* __restrict__ c0,             // [B,H]
    const float* __restrict__ pred_spd,       // [T,B]
    const float* __restrict__ W_se,           // [E,3]
    const float* __restrict__ b_se,           // [E]
    const float* __restrict__ W_ih,           // [4H,E]
    const float* __restrict__ W_hh,           // [4H,H]
    const float* __restrict__ b_ih,           // [4H]
    const float* __restrict__ b_hh,           // [4H]
    const float* __restrict__ W_hp,           // [2,H]
    const float* __restrict__ b_hp,           // [2]
    float* __restrict__ out) {                // [T*B*2] rels ++ [B*H] h
  extern __shared__ char smem[];
  _Float16* wt     = (_Float16*)smem;
  _Float16* xhbase = (_Float16*)(smem + WT_BYTES);
  float*    blstm  = (float*)(smem + WT_BYTES + XH_BYTES);
  float*    wse4   = blstm + NG;
  float*    whp    = wse4 + EE * 4;
  float*    bhp    = whp + 2 * HH;
  float*    relbuf = bhp + 4;

  const int tid  = threadIdx.x;
  const int wave = tid >> 5;
  const int lane = tid & 31;
  const int lo   = lane & 15;
  const int hi   = lane >> 4;
  const int gm0  = blockIdx.x * ROWS_PER_WG + wave * ROWS_PER_WAVE;

  __builtin_prefetch(W_hh, 0, 1);  // global_prefetch_b8: warm L2 path for weights

  // ---- one-time staging: f16 weights [n][k], fused bias, small mats ----
  for (int i = tid; i < NG * EE; i += THREADS) {
    int n = i >> 6, k = i & 63;
    wt[n * KP + k] = (_Float16)W_ih[i];
  }
  for (int i = tid; i < NG * HH; i += THREADS) {
    int n = i >> 7, k = i & 127;
    wt[n * KP + EE + k] = (_Float16)W_hh[i];
  }
  for (int i = tid; i < NG; i += THREADS) blstm[i] = b_ih[i] + b_hh[i];
  for (int i = tid; i < EE; i += THREADS) {
    wse4[i * 4 + 0] = W_se[i * 3 + 0];
    wse4[i * 4 + 1] = W_se[i * 3 + 1];
    wse4[i * 4 + 2] = W_se[i * 3 + 2];
    wse4[i * 4 + 3] = b_se[i];
  }
  for (int i = tid; i < 2 * HH; i += THREADS) whp[i] = W_hp[i];
  if (tid < 2) bhp[tid] = b_hp[tid];
  __syncthreads();

  // ---- per-wave persistent state ----
  _Float16* xh0 = xhbase + wave * 2 * (ROWS_PER_WAVE * KP);
  _Float16* xh1 = xh0 + ROWS_PER_WAVE * KP;
  float*    rb  = relbuf + wave * 64;

  // h0 -> f16 A-region of xh0 (cols 64..191)
  {
    int m = lo, cb = hi * 64, gm = gm0 + m;
#pragma unroll 8
    for (int c = 0; c < 64; ++c)
      xh0[m * KP + EE + cb + c] = (_Float16)h0[gm * HH + cb + c];
  }
  // c0 -> f32 accumulator (C/D) layout: VGPR r holds (M = r + 8*hi, N = hc*16 + lo)
  float c_st[64];
#pragma unroll
  for (int hc = 0; hc < 8; ++hc)
#pragma unroll
    for (int r = 0; r < 8; ++r)
      c_st[hc * 8 + r] = c0[(gm0 + r + hi * 8) * HH + hc * 16 + lo];

  // x0 = embed(last_pos_rel, speed_0)
  if (lane < 16) {
    int gm = gm0 + lane;
    rb[lane * 4 + 0] = last_pos_rel[gm * 2 + 0];
    rb[lane * 4 + 1] = last_pos_rel[gm * 2 + 1];
    rb[lane * 4 + 2] = pred_spd[gm];
  }
  embed_x(xh0, rb, wse4, lane);

  float* out_rels = out;
  float* out_h    = out + (size_t)TT * BB * 2;

  _Float16* cur = xh0;
  _Float16* nxt = xh1;

  // loop-invariant zero accumulator (hoisted once; no per-tile bias broadcast movs)
  v8f zacc;
#pragma unroll
  for (int r = 0; r < 8; ++r) zacc[r] = 0.0f;

  for (int t = 0; t < TT; ++t) {
    // gates[16, 512] via 8 H-chunks x 4 gates x 6 K-steps of v_wmma_f32_16x16x32_f16
#pragma unroll
    for (int hc = 0; hc < 8; ++hc) {
      v8f ai = zacc, af = zacc, ag = zacc, ao = zacc;
#pragma unroll
      for (int kt = 0; kt < 6; ++kt) {
        v16h a  = load_frag_a(cur, lane, kt);
        v16h b0 = load_frag_b(wt, lane, kt, 0 * 8 + hc);
        ai = WMMA_F16(a, b0, ai);
        v16h b1 = load_frag_b(wt, lane, kt, 1 * 8 + hc);
        af = WMMA_F16(a, b1, af);
        v16h b2 = load_frag_b(wt, lane, kt, 2 * 8 + hc);
        ag = WMMA_F16(a, b2, ag);
        v16h b3 = load_frag_b(wt, lane, kt, 3 * 8 + hc);
        ao = WMMA_F16(a, b3, ao);
      }
      // bias (pre-halved for sigmoid args) + LSTM cell update; h_{t+1} -> NEXT buffer
      float bi2 = 0.5f * blstm[0 * HH + hc * 16 + lo];
      float bf2 = 0.5f * blstm[1 * HH + hc * 16 + lo];
      float bg  =        blstm[2 * HH + hc * 16 + lo];
      float bo2 = 0.5f * blstm[3 * HH + hc * 16 + lo];
#pragma unroll
      for (int r = 0; r < 8; ++r) {
        float iv = sig_half_(__builtin_fmaf(0.5f, ai[r], bi2));
        float fv = sig_half_(__builtin_fmaf(0.5f, af[r], bf2));
        float gv = tanh_(ag[r] + bg);
        float ov = sig_half_(__builtin_fmaf(0.5f, ao[r], bo2));
        float cn = fv * c_st[hc * 8 + r] + iv * gv;
        float hn = ov * tanh_(cn);
        c_st[hc * 8 + r] = cn;
        nxt[(r + hi * 8) * KP + EE + hc * 16 + lo] = (_Float16)hn;
      }
    }

    // hidden2pos GEMV, full wave: lanes 0..15 -> rel.x of row m, lanes 16..31 -> rel.y
    {
      int m = lo, gm = gm0 + m, row = hi;  // row selects output column 0/1
      float racc = bhp[row];
      const float* wr = whp + row * HH;
#pragma unroll
      for (int k = 0; k < HH; k += 8) {
        v8h hv = *(const v8h*)(nxt + m * KP + EE + k);
#pragma unroll
        for (int j = 0; j < 8; ++j) racc = __builtin_fmaf((float)hv[j], wr[k + j], racc);
      }
      out_rels[((size_t)t * BB + gm) * 2 + row] = racc;
      rb[m * 4 + row] = racc;
      if (lane < 16) {
        int tn = (t + 1 < TT) ? (t + 1) : (TT - 1);
        rb[m * 4 + 2] = pred_spd[tn * BB + gm];
      }
    }
    embed_x(nxt, rb, wse4, lane);  // x_{t+1} into next buffer

    _Float16* tmp = cur; cur = nxt; nxt = tmp;
  }

  // final hidden state h_T (f16 in cur) -> f32 out
  {
    int m = lo, cb = hi * 64, gm = gm0 + m;
#pragma unroll 8
    for (int c = 0; c < 64; ++c)
      out_h[(size_t)gm * HH + cb + c] = (float)cur[m * KP + EE + cb + c];
  }
}

extern "C" void kernel_launch(void* const* d_in, const int* in_sizes, int n_in,
                              void* d_out, int out_size, void* d_ws, size_t ws_size,
                              hipStream_t stream) {
  (void)in_sizes; (void)n_in; (void)d_ws; (void)ws_size; (void)out_size;
  const float* last_pos     = (const float*)d_in[0];
  const float* last_pos_rel = (const float*)d_in[1];
  const float* h0           = (const float*)d_in[2];
  const float* c0           = (const float*)d_in[3];
  const float* pred_spd     = (const float*)d_in[4];
  const float* W_se         = (const float*)d_in[5];
  const float* b_se         = (const float*)d_in[6];
  const float* W_ih         = (const float*)d_in[7];
  const float* W_hh         = (const float*)d_in[8];
  const float* b_ih         = (const float*)d_in[9];
  const float* b_hh         = (const float*)d_in[10];
  const float* W_hp         = (const float*)d_in[11];
  const float* b_hp         = (const float*)d_in[12];

  (void)hipFuncSetAttribute((const void*)decoder_lstm_wmma,
                            hipFuncAttributeMaxDynamicSharedMemorySize,
                            (int)SMEM_BYTES);

  decoder_lstm_wmma<<<dim3(NUM_WG), dim3(THREADS), SMEM_BYTES, stream>>>(
      last_pos, last_pos_rel, h0, c0, pred_spd, W_se, b_se, W_ih, W_hh, b_ih,
      b_hh, W_hp, b_hp, (float*)d_out);
}
